// SemiPermeableAttention_1717986919046
// MI455X (gfx1250) — matrix operand order repaired and checked by reading, not compiled, not run on hardware
//
#include <hip/hip_runtime.h>
#include <hip/hip_bf16.h>
#include <stdint.h>

typedef __attribute__((ext_vector_type(16))) __bf16 v16bf;
typedef __attribute__((ext_vector_type(8)))  float  v8f;

union FragB16 {
  v16bf v;
  uint4 q[2];
};

__device__ __forceinline__ unsigned short f2bf(float f) {
  union { float f; uint32_t u; } c; c.f = f;
  uint32_t u = c.u + 0x7FFFu + ((c.u >> 16) & 1u);   // round-to-nearest-even
  return (unsigned short)(u >> 16);
}

// ---------------------------------------------------------------------------
// Elementwise f32 -> bf16
__global__ void cvt_bf16_kernel(const float* __restrict__ in,
                                unsigned short* __restrict__ out, int n) {
  int i = blockIdx.x * 256 + threadIdx.x;
  if (i < n) out[i] = f2bf(in[i]);
}

// in: [K][N] f32 row-major  ->  out: [N][K] bf16 (N-major, contiguous along K)
__global__ void transpose_bf16_kernel(const float* __restrict__ in,
                                      unsigned short* __restrict__ out,
                                      int K, int N) {
  int i = blockIdx.x * 256 + threadIdx.x;
  if (i < K * N) {
    int n = i / K, k = i - n * K;
    out[(size_t)n * K + k] = f2bf(in[(size_t)k * N + n]);
  }
}

// ---------------------------------------------------------------------------
// QKV GEMM: C[8192,3072] = Abf[8192,1024] * (BT[3072,1024])^T
// Epilogue scatters Q (scaled by 1/8), K as [bh][s][64], V^T as [bh][64][s], bf16.
__global__ __launch_bounds__(256) void qkv_gemm_kernel(
    const unsigned short* __restrict__ A,
    const unsigned short* __restrict__ BT,
    unsigned short* __restrict__ Qb,
    unsigned short* __restrict__ Kb,
    unsigned short* __restrict__ VTb) {
  constexpr int Kdim = 1024, LDA = 40, LDB = 40;  // 32 + 8 pad (bf16 elems)
  __shared__ __align__(16) unsigned short sA[128 * LDA];
  __shared__ __align__(16) unsigned short sB[128 * LDB];

  const int tid  = threadIdx.x;
  const int lane = tid & 31, r = lane & 15, half = lane >> 4;
  const int wave = tid >> 5;
  const int wm = (wave & 3) * 32;     // wave's 32-row strip in 128-row tile
  const int wn = (wave >> 2) * 64;    // wave's 64-col strip in 128-col tile
  const int blockN = blockIdx.x * 128;
  const int blockM = blockIdx.y * 128;

  const v8f vzero = {0.f, 0.f, 0.f, 0.f, 0.f, 0.f, 0.f, 0.f};
  v8f acc[2][4];
#pragma unroll
  for (int i = 0; i < 2; ++i)
#pragma unroll
    for (int j = 0; j < 4; ++j) acc[i][j] = vzero;

  for (int k0 = 0; k0 < Kdim; k0 += 32) {
#pragma unroll
    for (int t = 0; t < 2; ++t) {       // 512 x 16B vectors, 2 per thread
      int i = tid + t * 256;
      int row = i >> 2, seg = (i & 3) * 8;
      *(uint4*)&sA[row * LDA + seg] =
          *(const uint4*)&A[(size_t)(blockM + row) * Kdim + k0 + seg];
      *(uint4*)&sB[row * LDB + seg] =
          *(const uint4*)&BT[(size_t)(blockN + row) * Kdim + k0 + seg];
    }
    __syncthreads();

    FragB16 aF[2], bF[4];
#pragma unroll
    for (int mt = 0; mt < 2; ++mt) {    // A frag: lane row = r, K slices by half
      const char* p = (const char*)&sA[(wm + mt * 16 + r) * LDA];
      aF[mt].q[0] = *(const uint4*)(p + half * 16);
      aF[mt].q[1] = *(const uint4*)(p + 32 + half * 16);
    }
#pragma unroll
    for (int nt = 0; nt < 4; ++nt) {    // B frag: lane col = r, K slices by half
      const char* p = (const char*)&sB[(wn + nt * 16 + r) * LDB];
      bF[nt].q[0] = *(const uint4*)(p + half * 16);
      bF[nt].q[1] = *(const uint4*)(p + 32 + half * 16);
    }
#pragma unroll
    for (int mt = 0; mt < 2; ++mt)
#pragma unroll
      for (int nt = 0; nt < 4; ++nt)
        acc[mt][nt] = __builtin_amdgcn_wmma_f32_16x16x32_bf16(
            false, aF[mt].v, false, bF[nt].v, (short)0, acc[mt][nt], false, false);
    __syncthreads();
  }

#pragma unroll
  for (int mt = 0; mt < 2; ++mt)
#pragma unroll
    for (int nt = 0; nt < 4; ++nt)
#pragma unroll
      for (int v = 0; v < 8; ++v) {
        int m = blockM + wm + mt * 16 + half * 8 + v;
        int n = blockN + wn + nt * 16 + r;
        int b = m >> 11, s = m & 2047;
        int sec = n >> 10, nn = n & 1023;
        int h = nn >> 6, d = nn & 63;
        int bh = b * 16 + h;
        float val = acc[mt][nt][v];
        if (sec == 0)      Qb[((size_t)bh * 2048 + s) * 64 + d] = f2bf(val * 0.125f);
        else if (sec == 1) Kb[((size_t)bh * 2048 + s) * 64 + d] = f2bf(val);
        else               VTb[((size_t)bh * 64 + d) * 2048 + s] = f2bf(val);
      }
}

// ---------------------------------------------------------------------------
// Flash attention: grid (16 m-blocks of 128 rows, 64 bh). 8 waves x 16 q-rows.
__global__ __launch_bounds__(256) void attn_kernel(
    const unsigned short* __restrict__ Qb,
    const unsigned short* __restrict__ Kb,
    const unsigned short* __restrict__ VTb,
    unsigned short* __restrict__ AO) {
  constexpr int LDK = 72;  // 64 + 8 pad
  __shared__ __align__(16) unsigned short sK[64 * LDK];      // [key][dh]
  __shared__ __align__(16) unsigned short sV[64 * LDK];      // [dh][key]
  __shared__ __align__(16) unsigned short sP[8][16 * LDK];   // per-wave P

  const int tid  = threadIdx.x;
  const int lane = tid & 31, r = lane & 15, half = lane >> 4;
  const int wave = tid >> 5;
  const int bh = blockIdx.y;
  const int qm = blockIdx.x * 128 + wave * 16;   // first q row of this wave

  // Q fragments (A operand), dh=64 -> 2 WMMA k-steps, kept in registers
  FragB16 qF[2];
  {
    const unsigned short* qrow = &Qb[((size_t)bh * 2048 + qm + r) * 64];
#pragma unroll
    for (int ks = 0; ks < 2; ++ks) {
      qF[ks].q[0] = *(const uint4*)&qrow[ks * 32 + half * 8];
      qF[ks].q[1] = *(const uint4*)&qrow[ks * 32 + 16 + half * 8];
    }
  }

  const v8f vzero = {0.f, 0.f, 0.f, 0.f, 0.f, 0.f, 0.f, 0.f};
  v8f o[4];
#pragma unroll
  for (int dt = 0; dt < 4; ++dt) o[dt] = vzero;
  float mrow[8], lrow[8];
#pragma unroll
  for (int v = 0; v < 8; ++v) { mrow[v] = -1e30f; lrow[v] = 0.f; }

  const int nj = 2 * (blockIdx.x + 1);           // causal: only needed key blocks
  for (int j = 0; j < nj; ++j) {
    const int js = j * 64;
#pragma unroll
    for (int t = 0; t < 2; ++t) {                // cooperative K/V tile load
      int i = tid + t * 256;
      int row = i >> 3, seg = (i & 7) * 8;
      *(uint4*)&sK[row * LDK + seg] =
          *(const uint4*)&Kb[((size_t)bh * 2048 + js + row) * 64 + seg];
      *(uint4*)&sV[row * LDK + seg] =
          *(const uint4*)&VTb[((size_t)bh * 64 + row) * 2048 + js + seg];
    }
    __syncthreads();

    if (js <= qm + 15) {                         // wave-uniform => EXEC all-1s
      v8f s[4];
#pragma unroll
      for (int nt = 0; nt < 4; ++nt) s[nt] = vzero;
#pragma unroll
      for (int ks = 0; ks < 2; ++ks)
#pragma unroll
        for (int nt = 0; nt < 4; ++nt) {
          FragB16 kF;
          const char* p = (const char*)&sK[(nt * 16 + r) * LDK];
          kF.q[0] = *(const uint4*)(p + ks * 64 + half * 16);
          kF.q[1] = *(const uint4*)(p + ks * 64 + 32 + half * 16);
          s[nt] = __builtin_amdgcn_wmma_f32_16x16x32_bf16(
              false, qF[ks].v, false, kF.v, (short)0, s[nt], false, false);
        }

      if (js + 63 > qm) {                        // diagonal: additive soft mask
#pragma unroll
        for (int nt = 0; nt < 4; ++nt)
#pragma unroll
          for (int v = 0; v < 8; ++v)
            if (js + nt * 16 + r > qm + half * 8 + v) s[nt][v] += -1250.0f;
      }

      // online softmax; row stats live in the lane-half that owns the row
      float mnew[8], alpha[8], rsum[8];
#pragma unroll
      for (int v = 0; v < 8; ++v) {
        float x = fmaxf(fmaxf(s[0][v], s[1][v]), fmaxf(s[2][v], s[3][v]));
        x = fmaxf(x, __shfl_xor(x, 1));
        x = fmaxf(x, __shfl_xor(x, 2));
        x = fmaxf(x, __shfl_xor(x, 4));
        x = fmaxf(x, __shfl_xor(x, 8));
        mnew[v]  = fmaxf(mrow[v], x);
        alpha[v] = __expf(mrow[v] - mnew[v]);
        mrow[v]  = mnew[v];
        rsum[v]  = 0.f;
      }
#pragma unroll
      for (int nt = 0; nt < 4; ++nt)
#pragma unroll
        for (int v = 0; v < 8; ++v) {
          float p = __expf(s[nt][v] - mnew[v]);
          rsum[v] += p;
          sP[wave][(half * 8 + v) * LDK + nt * 16 + r] = f2bf(p);
        }
#pragma unroll
      for (int v = 0; v < 8; ++v) {
        float t = rsum[v];
        t += __shfl_xor(t, 1); t += __shfl_xor(t, 2);
        t += __shfl_xor(t, 4); t += __shfl_xor(t, 8);
        lrow[v] = lrow[v] * alpha[v] + t;
      }
#pragma unroll
      for (int dt = 0; dt < 4; ++dt)
#pragma unroll
        for (int v = 0; v < 8; ++v) o[dt][v] *= alpha[v];

      // in-wave LDS RAW: DS ops in-order per wave, just drain DScnt
      asm volatile("s_wait_dscnt 0" ::: "memory");

#pragma unroll
      for (int ks = 0; ks < 2; ++ks) {
        FragB16 pF;
        const char* pp = (const char*)&sP[wave][r * LDK];
        pF.q[0] = *(const uint4*)(pp + ks * 64 + half * 16);
        pF.q[1] = *(const uint4*)(pp + ks * 64 + 32 + half * 16);
#pragma unroll
        for (int dt = 0; dt < 4; ++dt) {
          FragB16 vF;
          const char* vp = (const char*)&sV[(dt * 16 + r) * LDK];
          vF.q[0] = *(const uint4*)(vp + ks * 64 + half * 16);
          vF.q[1] = *(const uint4*)(vp + ks * 64 + 32 + half * 16);
          o[dt] = __builtin_amdgcn_wmma_f32_16x16x32_bf16(
              false, pF.v, false, vF.v, (short)0, o[dt], false, false);
        }
      }
    }
    __syncthreads();
  }

  const int b = bh >> 4, h = bh & 15;
#pragma unroll
  for (int dt = 0; dt < 4; ++dt)
#pragma unroll
    for (int v = 0; v < 8; ++v) {
      int row = qm + half * 8 + v;
      float val = o[dt][v] / lrow[v];
      AO[((size_t)b * 2048 + row) * 1024 + h * 64 + dt * 16 + r] = f2bf(val);
    }
}

// ---------------------------------------------------------------------------
// Output GEMM: out[8192,1024] = AObf[8192,1024] * (woutT[1024,1024])^T + bias
__global__ __launch_bounds__(256) void out_gemm_kernel(
    const unsigned short* __restrict__ A,
    const unsigned short* __restrict__ BT,
    const float* __restrict__ bias,
    float* __restrict__ C) {
  constexpr int Kdim = 1024, LDA = 40, LDB = 40;
  __shared__ __align__(16) unsigned short sA[128 * LDA];
  __shared__ __align__(16) unsigned short sB[128 * LDB];

  const int tid  = threadIdx.x;
  const int lane = tid & 31, r = lane & 15, half = lane >> 4;
  const int wave = tid >> 5;
  const int wm = (wave & 3) * 32;
  const int wn = (wave >> 2) * 64;
  const int blockN = blockIdx.x * 128;
  const int blockM = blockIdx.y * 128;

  const v8f vzero = {0.f, 0.f, 0.f, 0.f, 0.f, 0.f, 0.f, 0.f};
  v8f acc[2][4];
#pragma unroll
  for (int i = 0; i < 2; ++i)
#pragma unroll
    for (int j = 0; j < 4; ++j) acc[i][j] = vzero;

  for (int k0 = 0; k0 < Kdim; k0 += 32) {
#pragma unroll
    for (int t = 0; t < 2; ++t) {
      int i = tid + t * 256;
      int row = i >> 2, seg = (i & 3) * 8;
      *(uint4*)&sA[row * LDA + seg] =
          *(const uint4*)&A[(size_t)(blockM + row) * Kdim + k0 + seg];
      *(uint4*)&sB[row * LDB + seg] =
          *(const uint4*)&BT[(size_t)(blockN + row) * Kdim + k0 + seg];
    }
    __syncthreads();

    FragB16 aF[2], bF[4];
#pragma unroll
    for (int mt = 0; mt < 2; ++mt) {
      const char* p = (const char*)&sA[(wm + mt * 16 + r) * LDA];
      aF[mt].q[0] = *(const uint4*)(p + half * 16);
      aF[mt].q[1] = *(const uint4*)(p + 32 + half * 16);
    }
#pragma unroll
    for (int nt = 0; nt < 4; ++nt) {
      const char* p = (const char*)&sB[(wn + nt * 16 + r) * LDB];
      bF[nt].q[0] = *(const uint4*)(p + half * 16);
      bF[nt].q[1] = *(const uint4*)(p + 32 + half * 16);
    }
#pragma unroll
    for (int mt = 0; mt < 2; ++mt)
#pragma unroll
      for (int nt = 0; nt < 4; ++nt)
        acc[mt][nt] = __builtin_amdgcn_wmma_f32_16x16x32_bf16(
            false, aF[mt].v, false, bF[nt].v, (short)0, acc[mt][nt], false, false);
    __syncthreads();
  }

#pragma unroll
  for (int mt = 0; mt < 2; ++mt)
#pragma unroll
    for (int nt = 0; nt < 4; ++nt)
#pragma unroll
      for (int v = 0; v < 8; ++v) {
        int m = blockM + wm + mt * 16 + half * 8 + v;
        int n = blockN + wn + nt * 16 + r;
        C[(size_t)m * 1024 + n] = acc[mt][nt][v] + bias[n];
      }
}

// ---------------------------------------------------------------------------
extern "C" void kernel_launch(void* const* d_in, const int* in_sizes, int n_in,
                              void* d_out, int out_size, void* d_ws, size_t ws_size,
                              hipStream_t stream) {
  const float* x     = (const float*)d_in[0];
  const float* w_qkv = (const float*)d_in[1];
  const float* w_out = (const float*)d_in[2];
  const float* b_out = (const float*)d_in[3];
  float* out = (float*)d_out;

  // workspace layout (bf16), ~88 MiB total — fits in MI455X's 192 MB L2
  unsigned short* x_bf  = (unsigned short*)d_ws;
  unsigned short* wqkvT = x_bf  + (size_t)8192 * 1024;
  unsigned short* woutT = wqkvT + (size_t)3072 * 1024;
  unsigned short* Qb    = woutT + (size_t)1024 * 1024;
  unsigned short* Kb    = Qb    + (size_t)64 * 2048 * 64;
  unsigned short* VTb   = Kb    + (size_t)64 * 2048 * 64;
  unsigned short* AOb   = VTb   + (size_t)64 * 64 * 2048;

  cvt_bf16_kernel<<<(8192 * 1024) / 256, 256, 0, stream>>>(x, x_bf, 8192 * 1024);
  transpose_bf16_kernel<<<(1024 * 3072) / 256, 256, 0, stream>>>(w_qkv, wqkvT, 1024, 3072);
  transpose_bf16_kernel<<<(1024 * 1024) / 256, 256, 0, stream>>>(w_out, woutT, 1024, 1024);
  qkv_gemm_kernel<<<dim3(24, 64), 256, 0, stream>>>(x_bf, wqkvT, Qb, Kb, VTb);
  attn_kernel<<<dim3(16, 64), 256, 0, stream>>>(Qb, Kb, VTb, AOb);
  out_gemm_kernel<<<dim3(8, 64), 256, 0, stream>>>(AOb, woutT, b_out, out);
}
